// DAWN_45526653337606
// MI455X (gfx1250) — compile-verified, matrix operand order checked by reading, not run on hardware
//
#include <hip/hip_runtime.h>
#include <hip/hip_bf16.h>

#define BB      2
#define SS      1024
#define DD      512
#define NN      4096
#define NCELLS  16
#define MPC     32
#define NCAND   288     // 9 * 32
#define MAXK    32
#define CHUNKC  128
#define NCHUNK  8       // SS / CHUNKC
#define NHEADS  8
#define DH      64      // DD / NHEADS

typedef __attribute__((ext_vector_type(16))) __bf16         v16bf;
typedef __attribute__((ext_vector_type(16))) unsigned short v16us;
typedef __attribute__((ext_vector_type(8)))  float          v8f;
typedef __attribute__((ext_vector_type(4)))  unsigned int   v4u;
typedef __attribute__((ext_vector_type(8)))  int            v8i;
typedef __attribute__((ext_vector_type(4)))  int            v4i;

union F16frag { v16us u; v16bf b; };

// ---------------- helpers ----------------

__device__ __forceinline__ unsigned short f2bf(float f) {
    unsigned int u = __float_as_uint(f);
    u += 0x7fffu + ((u >> 16) & 1u);     // round-to-nearest-even
    return (unsigned short)(u >> 16);
}

// Tensor Data Mover: DMA a (tile_h x tile_w) bf16 tile (row stride `row_stride`
// elements) from global memory into contiguous row-major LDS. D# per CDNA5 ISA §8.
// 6-arg builtin form: (g0: u32x4, g1: i32x8, g2: i32x4, g3: i32x4, i32x8, cpol)
__device__ __forceinline__ void tdm_load_2d(unsigned lds_addr, const void* gaddr,
                                            unsigned tile_w, unsigned tile_h,
                                            unsigned row_stride)
{
    unsigned long long ga = (unsigned long long)(size_t)gaddr;
    v4u g0;
    g0[0] = 1u;                                   // count=1 (valid user descriptor)
    g0[1] = lds_addr;                             // lds_addr[31:0]   (bits 63:32)
    g0[2] = (unsigned)ga;                         // global_addr[31:0] (bits 95:64)
    g0[3] = (unsigned)(ga >> 32) | (2u << 30);    // global_addr[56:32] | type=2
    v8i g1;
    g1[0] = (int)(1u << 16);                      // workgroup_mask=0 | data_size=1 (2B)
    g1[1] = (int)(tile_w << 16);                  // abar=0 | tensor_dim0[15:0]
    g1[2] = (int)(tile_h << 16);                  // tensor_dim0[31:16]=0 | tensor_dim1[15:0]
    g1[3] = (int)(tile_w << 16);                  // tensor_dim1[31:16]=0 | tile_dim0
    g1[4] = (int)tile_h;                          // tile_dim1 | tile_dim2=0
    g1[5] = (int)row_stride;                      // tensor_dim0_stride[31:0]
    g1[6] = 0;                                    // stride hi | tensor_dim1_stride lo
    g1[7] = 0;
    v4i z4 = {0, 0, 0, 0};                        // 2-D tensor: groups 2/3 unused
    v8i z8 = {0, 0, 0, 0, 0, 0, 0, 0};
    __builtin_amdgcn_tensor_load_to_lds(g0, g1, z4, z4, z8, 0);
}

__device__ __forceinline__ unsigned lds_off(const void* p) {
    // generic pointers to LDS carry the byte offset in the low 32 bits (ISA §10.2)
    return (unsigned)(size_t)p;
}

__device__ __forceinline__ float block_reduce_sum256(float v, float* red, int tid) {
    red[tid] = v; __syncthreads();
    for (int st = 128; st > 0; st >>= 1) {
        if (tid < st) red[tid] += red[tid + st];
        __syncthreads();
    }
    float r = red[0]; __syncthreads();
    return r;
}

__device__ __forceinline__ float block_reduce_max256(float v, float* red, int tid) {
    red[tid] = v; __syncthreads();
    for (int st = 128; st > 0; st >>= 1) {
        if (tid < st) red[tid] = fmaxf(red[tid], red[tid + st]);
        __syncthreads();
    }
    float r = red[0]; __syncthreads();
    return r;
}

// threshold_gate: eg = exp(gate)-1; keep top-32 (T = 32nd largest), normalize, * tanh(max)
__device__ __forceinline__ void gate_block(const float* act, const unsigned char* valid,
                                           float tau, float* eg, float* g, float* red, int tid)
{
    for (int i = tid; i < NCAND; i += 256) {
        float s    = valid[i] ? act[i] : -1e9f;
        float raw  = s - tau;
        float gate = raw > 0.f ? raw : 1e-8f * expf(raw);
        eg[i] = expf(gate) - 1.f;
    }
    __syncthreads();
    // T = max{ v : #{j: eg[j] >= v} >= 32 }  ==  32nd-largest value (tie-exact)
    float localT = -1.f;
    for (int i = tid; i < NCAND; i += 256) {
        float v = eg[i];
        int cnt = 0;
        for (int j = 0; j < NCAND; ++j) cnt += (eg[j] >= v) ? 1 : 0;
        if (cnt >= MAXK) localT = fmaxf(localT, v);
    }
    float T = block_reduce_max256(localT, red, tid);
    float lsum = 0.f, lmax = -1e30f;
    for (int i = tid; i < NCAND; i += 256) {
        float v    = eg[i];
        float kept = (v >= T) ? v : 0.f;
        g[i] = kept;
        lsum += kept;
        lmax = fmaxf(lmax, v);
    }
    float gs = block_reduce_sum256(lsum, red, tid) + 1e-8f;
    float mx = block_reduce_max256(lmax, red, tid);
    float scale = tanhf(mx) / gs;
    for (int i = tid; i < NCAND; i += 256) g[i] *= scale;
    __syncthreads();
}

// out[d] = sum_n (act*g*valid)[n] * neurons[idx[n]][d]  via compacted nonzero list
__device__ __forceinline__ void accum_out(const float* act, const unsigned char* valid,
                                          const int* idx, const float* g,
                                          const float* __restrict__ neurons,
                                          unsigned short* __restrict__ dst,
                                          float* wcomp, int* ncomp, int* cnt_nz, int tid)
{
    if (tid == 0) *cnt_nz = 0;
    __syncthreads();
    for (int i = tid; i < NCAND; i += 256) {
        if (valid[i] && g[i] != 0.f) {
            int p = atomicAdd(cnt_nz, 1);
            ncomp[p] = idx[i];
            wcomp[p] = act[i] * g[i];
        }
    }
    __syncthreads();
    int nnz = *cnt_nz;
    float a0 = 0.f, a1 = 0.f;
    for (int p = 0; p < nnz; ++p) {
        float w = wcomp[p];
        const float* row = neurons + (size_t)ncomp[p] * DD;
        a0 += w * row[tid];
        a1 += w * row[tid + 256];
    }
    dst[tid]       = f2bf(a0);
    dst[tid + 256] = f2bf(a1);
    __syncthreads();
}

__device__ __forceinline__ void accum_loss(const unsigned char* valid, const int* idx,
                                           const float* g, const float* __restrict__ npos,
                                           float px, float py, float* red, int tid,
                                           float* sum_slot, float* cnt_slot)
{
    float lpl = 0.f, lcnt = 0.f;
    for (int i = tid; i < NCAND; i += 256) {
        if (valid[i]) {
            lcnt += 1.f;
            if (g[i] != 0.f) {
                float dx = px - npos[idx[i] * 2 + 0];
                float dy = py - npos[idx[i] * 2 + 1];
                lpl += g[i] * (dx * dx + dy * dy);
            }
        }
    }
    float tot = block_reduce_sum256(lpl, red, tid);
    float cnt = block_reduce_sum256(lcnt, red, tid);
    if (tid == 0) { atomicAdd(sum_slot, tot); atomicAdd(cnt_slot, cnt); }
}

// ---------------- kernel 0: expand_O -> bf16, zero loss accumulators ----------------

__global__ __launch_bounds__(256) void prep_kernel(const float* __restrict__ W,
                                                   unsigned short* __restrict__ Wbf,
                                                   float* __restrict__ loss_acc)
{
    int i = blockIdx.x * 256 + threadIdx.x;
    Wbf[i] = f2bf(W[i]);
    if (blockIdx.x == 0 && threadIdx.x < 4 * NCHUNK) loss_acc[threadIdx.x] = 0.f;
}

// ---------------- kernel 1: per-token candidate gather + gating -> Q/K/V (bf16) ----------------

__global__ __launch_bounds__(256) void qkv_kernel(
    const float* __restrict__ x, const float* __restrict__ qk_pos, const float* __restrict__ v_pos,
    const float* __restrict__ tauQ, const float* __restrict__ tauK, const float* __restrict__ tauV,
    const float* __restrict__ qk_neurons, const float* __restrict__ v_neurons,
    const float* __restrict__ npos_qk, const float* __restrict__ npos_v,
    const int* __restrict__ cmap_qk, const int* __restrict__ cmap_v,
    const float* __restrict__ pos_min, const float* __restrict__ pos_range,
    unsigned short* __restrict__ Qbf, unsigned short* __restrict__ Kbf,
    unsigned short* __restrict__ Vbf, float* __restrict__ loss_acc)
{
    const int t    = blockIdx.x;            // token in [0, B*S)
    const int s    = t % SS;
    const int tid  = threadIdx.x;
    const int lane = tid & 31;
    const int wv   = tid >> 5;
    const int chunk = s / CHUNKC;

    __shared__ float         xs[DD];
    __shared__ int           idxA[NCAND], idxB[NCAND];
    __shared__ unsigned char valA[NCAND], valB[NCAND];
    __shared__ float         actA[NCAND], actB[NCAND];
    __shared__ float         eg[NCAND], g[NCAND];
    __shared__ float         red[256];
    __shared__ float         wcomp[NCAND];
    __shared__ int           ncomp[NCAND];
    __shared__ int           cnt_nz;

    xs[tid]       = x[(size_t)t * DD + tid];
    xs[tid + 256] = x[(size_t)t * DD + tid + 256];

    const float pm0 = pos_min[0], pm1 = pos_min[1];
    const float pr0 = pos_range[0], pr1 = pos_range[1];
    const float qpx = qk_pos[t * 2 + 0], qpy = qk_pos[t * 2 + 1];
    const float vpx = v_pos[t * 2 + 0],  vpy = v_pos[t * 2 + 1];

    int cqx = (int)(((qpx - pm0) / pr0) * (float)NCELLS); cqx = min(max(cqx, 0), NCELLS - 1);
    int cqy = (int)(((qpy - pm1) / pr1) * (float)NCELLS); cqy = min(max(cqy, 0), NCELLS - 1);
    int cvx = (int)(((vpx - pm0) / pr0) * (float)NCELLS); cvx = min(max(cvx, 0), NCELLS - 1);
    int cvy = (int)(((vpy - pm1) / pr1) * (float)NCELLS); cvy = min(max(cvy, 0), NCELLS - 1);

    for (int i = tid; i < NCAND; i += 256) {
        int nb = i / MPC, slot = i - nb * MPC;
        int ox = nb / 3 - 1, oy = nb % 3 - 1;
        {
            int gx = min(max(cqx + ox, 0), NCELLS - 1);
            int gy = min(max(cqy + oy, 0), NCELLS - 1);
            int c  = cmap_qk[(gx * NCELLS + gy) * MPC + slot];
            valA[i] = (c >= 0); idxA[i] = (c >= 0) ? c : 0;
        }
        {
            int gx = min(max(cvx + ox, 0), NCELLS - 1);
            int gy = min(max(cvy + oy, 0), NCELLS - 1);
            int c  = cmap_v[(gx * NCELLS + gy) * MPC + slot];
            valB[i] = (c >= 0); idxB[i] = (c >= 0) ? c : 0;
        }
    }
    __syncthreads();

    // activations: one candidate per wave, lane-strided dot of length 512
    for (int i = wv; i < NCAND; i += 8) {
        const float* rowA = qk_neurons + (size_t)idxA[i] * DD;
        const float* rowB = v_neurons  + (size_t)idxB[i] * DD;
        if (i + 8 < NCAND) {
            __builtin_prefetch(qk_neurons + (size_t)idxA[i + 8] * DD + lane, 0, 0);
            __builtin_prefetch(v_neurons  + (size_t)idxB[i + 8] * DD + lane, 0, 0);
        }
        float pa = 0.f, pb = 0.f;
        #pragma unroll
        for (int e = 0; e < DD / 32; ++e) {
            float xv = xs[lane + 32 * e];
            pa += xv * rowA[lane + 32 * e];
            pb += xv * rowB[lane + 32 * e];
        }
        #pragma unroll
        for (int m = 16; m >= 1; m >>= 1) {
            pa += __shfl_xor(pa, m, 32);
            pb += __shfl_xor(pb, m, 32);
        }
        if (lane == 0) { actA[i] = pa; actB[i] = pb; }
    }
    __syncthreads();

    // gQ -> Q (+ qk pos loss)
    gate_block(actA, valA, tauQ[t], eg, g, red, tid);
    accum_loss(valA, idxA, g, npos_qk, qpx, qpy, red, tid,
               &loss_acc[chunk], &loss_acc[NCHUNK + chunk]);
    accum_out(actA, valA, idxA, g, qk_neurons, Qbf + (size_t)t * DD, wcomp, ncomp, &cnt_nz, tid);

    // gK -> K
    gate_block(actA, valA, tauK[t], eg, g, red, tid);
    accum_out(actA, valA, idxA, g, qk_neurons, Kbf + (size_t)t * DD, wcomp, ncomp, &cnt_nz, tid);

    // gV -> V (+ v pos loss)
    gate_block(actB, valB, tauV[t], eg, g, red, tid);
    accum_loss(valB, idxB, g, npos_v, vpx, vpy, red, tid,
               &loss_acc[2 * NCHUNK + chunk], &loss_acc[3 * NCHUNK + chunk]);
    accum_out(actB, valB, idxB, g, v_neurons, Vbf + (size_t)t * DD, wcomp, ncomp, &cnt_nz, tid);
}

// ---------------- kernel 2: causal flash attention, bf16 WMMA + TDM double buffering ----------------

__global__ __launch_bounds__(128) void attn_kernel(const unsigned short* __restrict__ Qbf,
                                                   const unsigned short* __restrict__ Kbf,
                                                   const unsigned short* __restrict__ Vbf,
                                                   unsigned short* __restrict__ Obf)
{
    const int mblk = blockIdx.x, h = blockIdx.y, b = blockIdx.z;
    const int tid  = threadIdx.x;
    const int lane = tid & 31, wv = tid >> 5;        // 4 waves
    const int hi   = lane >> 4, ln = lane & 15;

    __shared__ unsigned short Qt[64][DH];            // 8 KB
    __shared__ unsigned short Kt[2][16][DH];         // 2x2 KB double buffer
    __shared__ unsigned short Vt[2][16][DH];         // 2x2 KB double buffer
    __shared__ unsigned short Pt[4][16][32];         // per-wave P staging (K padded to 32)

    const int q0 = mblk * 64;
    const size_t base = (size_t)b * SS * DD + (size_t)h * DH;
    const unsigned short* Kg = Kbf + base;
    const unsigned short* Vg = Vbf + base;

    // TDM prologue: Q tile + first K/V tiles (single wave issues; TENSORcnt tracked per wave)
    if (wv == 0) {
        tdm_load_2d(lds_off(&Qt[0][0]),    Qbf + base + (size_t)q0 * DD, DH, 64, DD);
        tdm_load_2d(lds_off(&Kt[0][0][0]), Kg,                           DH, 16, DD);
        tdm_load_2d(lds_off(&Vt[0][0][0]), Vg,                           DH, 16, DD);
        __builtin_amdgcn_s_wait_tensorcnt(0);
    }
    // zero padded upper half (k = 16..31) of the per-wave P tiles
    for (int i = tid; i < 4 * 16 * 16; i += 128) {
        int w2 = i >> 8, m = (i >> 4) & 15, c = i & 15;
        Pt[w2][m][16 + c] = 0;
    }
    __syncthreads();

    // Q A-fragments from LDS (rows q0 + wv*16 + m, K split 0..31 / 32..63)
    F16frag qa0, qa1;
    #pragma unroll
    for (int e = 0; e < 16; ++e) {
        int k = (e & 7) + hi * 8 + ((e >= 8) ? 16 : 0);
        qa0.u[e] = Qt[wv * 16 + ln][k];
        qa1.u[e] = Qt[wv * 16 + ln][32 + k];
    }

    v8f oacc[4] = {};
    float mrow[8], lrow[8];
    #pragma unroll
    for (int r = 0; r < 8; ++r) { mrow[r] = -1e30f; lrow[r] = 0.f; }

    const int jmax = (q0 + 63) >> 4;
    for (int j = 0; j <= jmax; ++j) {
        const int buf = j & 1;
        // prefetch next K/V tiles into the other buffer; wait for current tiles
        if (wv == 0) {
            if (j < jmax) {
                tdm_load_2d(lds_off(&Kt[buf ^ 1][0][0]), Kg + (size_t)(j + 1) * 16 * DD, DH, 16, DD);
                tdm_load_2d(lds_off(&Vt[buf ^ 1][0][0]), Vg + (size_t)(j + 1) * 16 * DD, DH, 16, DD);
                __builtin_amdgcn_s_wait_tensorcnt(2);    // in-order per wave: current tiles done
            } else {
                __builtin_amdgcn_s_wait_tensorcnt(0);
            }
        }
        __syncthreads();

        // S = Q(16x64) * K^T(64x16):  B[k][n] = Kt[n][k]
        F16frag kb0, kb1;
        #pragma unroll
        for (int e = 0; e < 16; ++e) {
            int k = (e & 7) + hi * 8 + ((e >= 8) ? 16 : 0);
            kb0.u[e] = Kt[buf][ln][k];
            kb1.u[e] = Kt[buf][ln][32 + k];
        }
        v8f sc = {};
        sc = __builtin_amdgcn_wmma_f32_16x16x32_bf16(false, qa0.b, false, kb0.b, (short)0, sc, false, false);
        sc = __builtin_amdgcn_wmma_f32_16x16x32_bf16(false, qa1.b, false, kb1.b, (short)0, sc, false, false);

        // mask + scale + online softmax (rows live in 16-lane halves)
        float pv[8], alpha[8];
        #pragma unroll
        for (int r = 0; r < 8; ++r) {
            int q  = q0 + wv * 16 + r + 8 * hi;
            int kk = j * 16 + ln;
            float sv = sc[r] * 0.125f;                   // 1/sqrt(64)
            pv[r] = (kk <= q) ? sv : -1e30f;
        }
        #pragma unroll
        for (int r = 0; r < 8; ++r) {
            float v = pv[r];
            v = fmaxf(v, __shfl_xor(v, 1, 32));
            v = fmaxf(v, __shfl_xor(v, 2, 32));
            v = fmaxf(v, __shfl_xor(v, 4, 32));
            v = fmaxf(v, __shfl_xor(v, 8, 32));
            float mn = fmaxf(mrow[r], v);
            alpha[r] = __expf(mrow[r] - mn);
            mrow[r]  = mn;
            float p  = __expf(pv[r] - mn);
            pv[r]    = p;
            p += __shfl_xor(p, 1, 32);
            p += __shfl_xor(p, 2, 32);
            p += __shfl_xor(p, 4, 32);
            p += __shfl_xor(p, 8, 32);
            lrow[r] = lrow[r] * alpha[r] + p;
        }
        #pragma unroll
        for (int c = 0; c < 4; ++c)
            #pragma unroll
            for (int r = 0; r < 8; ++r) oacc[c][r] *= alpha[r];

        // stage P (16x16, zero-padded to K=32) through per-wave LDS
        #pragma unroll
        for (int r = 0; r < 8; ++r) Pt[wv][r + 8 * hi][ln] = f2bf(pv[r]);

        F16frag pa;
        #pragma unroll
        for (int e = 0; e < 16; ++e) {
            int k = (e & 7) + hi * 8 + ((e >= 8) ? 16 : 0);
            pa.u[e] = Pt[wv][ln][k];
        }
        #pragma unroll
        for (int c = 0; c < 4; ++c) {
            F16frag vb;
            #pragma unroll
            for (int e = 0; e < 16; ++e) {
                int k = (e & 7) + hi * 8 + ((e >= 8) ? 16 : 0);
                vb.u[e] = (k < 16) ? Vt[buf][k][c * 16 + ln] : (unsigned short)0;
            }
            oacc[c] = __builtin_amdgcn_wmma_f32_16x16x32_bf16(false, pa.b, false, vb.b, (short)0, oacc[c], false, false);
        }
        __syncthreads();   // all reads of Kt[buf]/Vt[buf] done before next DMA overwrites buf^1
    }

    #pragma unroll
    for (int c = 0; c < 4; ++c)
        #pragma unroll
        for (int r = 0; r < 8; ++r) {
            int q = q0 + wv * 16 + r + 8 * hi;
            Obf[base + (size_t)q * DD + c * 16 + ln] = f2bf(oacc[c][r] / lrow[r]);
        }
}

// ---------------- kernel 3: out = attn_out(bf16) @ expand_O(bf16) -> f32 (TDM pipelined) ----------------

__global__ __launch_bounds__(128) void proj_kernel(const unsigned short* __restrict__ A,
                                                   const unsigned short* __restrict__ W,
                                                   float* __restrict__ out)
{
    const int mb = blockIdx.x, nb = blockIdx.y;
    const int tid  = threadIdx.x;
    const int lane = tid & 31, wv = tid >> 5;
    const int hi   = lane >> 4, ln = lane & 15;

    __shared__ unsigned short At[2][64][32];
    __shared__ unsigned short Wt[2][32][64];

    const unsigned short* Ag = A + (size_t)(mb * 64) * DD;
    const unsigned short* Wg = W + nb * 64;

    if (wv == 0) {
        tdm_load_2d(lds_off(&At[0][0][0]), Ag, 32, 64, DD);
        tdm_load_2d(lds_off(&Wt[0][0][0]), Wg, 64, 32, DD);
    }

    v8f oacc[4] = {};
    for (int kk = 0; kk < DD / 32; ++kk) {
        const int buf = kk & 1;
        if (wv == 0) {
            if (kk < DD / 32 - 1) {
                tdm_load_2d(lds_off(&At[buf ^ 1][0][0]), Ag + (kk + 1) * 32, 32, 64, DD);
                tdm_load_2d(lds_off(&Wt[buf ^ 1][0][0]), Wg + (size_t)((kk + 1) * 32) * DD, 64, 32, DD);
                __builtin_amdgcn_s_wait_tensorcnt(2);
            } else {
                __builtin_amdgcn_s_wait_tensorcnt(0);
            }
        }
        __syncthreads();

        F16frag af;
        #pragma unroll
        for (int e = 0; e < 16; ++e) {
            int k = (e & 7) + hi * 8 + ((e >= 8) ? 16 : 0);
            af.u[e] = At[buf][wv * 16 + ln][k];
        }
        #pragma unroll
        for (int c = 0; c < 4; ++c) {
            F16frag bf_;
            #pragma unroll
            for (int e = 0; e < 16; ++e) {
                int k = (e & 7) + hi * 8 + ((e >= 8) ? 16 : 0);
                bf_.u[e] = Wt[buf][k][c * 16 + ln];
            }
            oacc[c] = __builtin_amdgcn_wmma_f32_16x16x32_bf16(false, af.b, false, bf_.b, (short)0, oacc[c], false, false);
        }
        __syncthreads();   // reads of buf complete before next DMA overwrites buf^1
    }

    #pragma unroll
    for (int c = 0; c < 4; ++c)
        #pragma unroll
        for (int r = 0; r < 8; ++r)
            out[(size_t)(mb * 64 + wv * 16 + r + 8 * hi) * DD + nb * 64 + c * 16 + ln] = oacc[c][r];
}

// ---------------- kernel 4: finalize pos_loss ----------------

__global__ void loss_kernel(const float* __restrict__ acc, float* __restrict__ out_loss)
{
    if (threadIdx.x == 0 && blockIdx.x == 0) {
        float s = 0.f;
        for (int c = 0; c < NCHUNK; ++c) {
            s += acc[c]              / (acc[NCHUNK + c]     + 1e-8f);
            s += acc[2 * NCHUNK + c] / (acc[3 * NCHUNK + c] + 1e-8f);
        }
        out_loss[0] = s / (float)NCHUNK;
    }
}

// ---------------- launch ----------------

extern "C" void kernel_launch(void* const* d_in, const int* in_sizes, int n_in,
                              void* d_out, int out_size, void* d_ws, size_t ws_size,
                              hipStream_t stream)
{
    const float* x        = (const float*)d_in[0];
    const float* qk_pos   = (const float*)d_in[1];
    const float* v_pos    = (const float*)d_in[2];
    const float* tau_Q    = (const float*)d_in[3];
    const float* tau_K    = (const float*)d_in[4];
    const float* tau_V    = (const float*)d_in[5];
    const float* qk_neur  = (const float*)d_in[6];
    const float* v_neur   = (const float*)d_in[7];
    const float* npos_qk  = (const float*)d_in[8];
    const float* npos_v   = (const float*)d_in[9];
    const int*   cmap_qk  = (const int*)d_in[10];
    const int*   cmap_v   = (const int*)d_in[11];
    const float* pos_min  = (const float*)d_in[12];
    const float* pos_rng  = (const float*)d_in[13];
    const float* expand_O = (const float*)d_in[14];

    const size_t nQ = (size_t)BB * SS * DD;              // 1,048,576 elements
    unsigned short* Qbf = (unsigned short*)d_ws;
    unsigned short* Kbf = Qbf + nQ;
    unsigned short* Vbf = Kbf + nQ;
    unsigned short* Obf = Vbf + nQ;
    unsigned short* Wbf = Obf + nQ;
    float* loss_acc = (float*)(Wbf + (size_t)DD * DD);   // 4*NCHUNK floats

    prep_kernel<<<dim3((DD * DD) / 256), 256, 0, stream>>>(expand_O, Wbf, loss_acc);

    qkv_kernel<<<dim3(BB * SS), 256, 0, stream>>>(
        x, qk_pos, v_pos, tau_Q, tau_K, tau_V, qk_neur, v_neur,
        npos_qk, npos_v, cmap_qk, cmap_v, pos_min, pos_rng,
        Qbf, Kbf, Vbf, loss_acc);

    attn_kernel<<<dim3(SS / 64, NHEADS, BB), 128, 0, stream>>>(Qbf, Kbf, Vbf, Obf);

    proj_kernel<<<dim3((BB * SS) / 64, DD / 64), 128, 0, stream>>>(Obf, Wbf, (float*)d_out);

    loss_kernel<<<1, 1, 0, stream>>>(loss_acc, (float*)d_out + nQ);
}
